// DLRM_RPC_21251498180656
// MI455X (gfx1250) — matrix operand order repaired and checked by reading, not compiled, not run on hardware
//
#include <hip/hip_runtime.h>
#include <hip/hip_bf16.h>
#include <math.h>

// ---------------------------------------------------------------------------
// DLRM forward for MI455X (gfx1250, wave32, WMMA f32_16x16x32_f16 + TDM)
// ---------------------------------------------------------------------------

typedef __attribute__((ext_vector_type(16))) _Float16 v16h;
typedef __attribute__((ext_vector_type(8)))  _Float16 v8h;
typedef __attribute__((ext_vector_type(4)))  _Float16 v4h;
typedef __attribute__((ext_vector_type(8)))  float    v8f;
typedef __attribute__((ext_vector_type(4)))  float    v4f;
typedef unsigned int __attribute__((ext_vector_type(4))) u32x4;
typedef int          __attribute__((ext_vector_type(4))) i32x4;
typedef int          __attribute__((ext_vector_type(8))) i32x8;

#define BATCH   16384
#define EMB_D   128
#define N_EMB   26
#define VOCAB   50000

#if __has_builtin(__builtin_amdgcn_tensor_load_to_lds)
#define HAVE_TDM 1
#endif

// ---------------------------------------------------------------------------
// WMMA fragment loaders (wave32; A and B both use contiguous k order, not the ISA table order)
//  A 16x32 f16 (MxK): lane l: row m=l&15, hl=l>>4;
//    element i -> K = kb + hl*16 + i                 => one 32B load.
//  B 32x16 f16 (KxN) stored N-major: lane l: col n=l&15, hl=l>>4;
//    element i -> K = kb + hl*16 + i                 => one 32B load.
//  C/D 16x16 f32: lane l: col=l&15; element j -> row=(l>>4)*8+j.
// ---------------------------------------------------------------------------
__device__ inline v16h load_a_frag(const _Float16* Abase, int lda, int kb, int lane) {
    const int m  = lane & 15;
    const int hl = lane >> 4;
    const _Float16* p = Abase + (size_t)m * lda + kb + hl * 16;
    return *(const v16h*)p;
}

__device__ inline v16h load_b_frag(const _Float16* Bbase, int ldk, int n0, int kb, int lane) {
    const int n  = lane & 15;
    const int hl = lane >> 4;
    return *(const v16h*)(Bbase + (size_t)(n0 + n) * ldk + kb + hl * 16);
}

// ---------------------------------------------------------------------------
// TDM: 2D tile load Global -> LDS via Tensor Data Mover (ISA ch.8 D# layout).
// Tile = klen (contiguous, elems) x rows; source row stride = ld elems (f16).
// LDS receives the tile packed contiguously (row r at lds_off + r*klen*2).
// ---------------------------------------------------------------------------
#ifdef HAVE_TDM
__device__ inline void tdm_load_2d(const _Float16* gsrc, unsigned lds_off,
                                   unsigned klen, unsigned rows, unsigned ld) {
    unsigned long long ga = (unsigned long long)(uintptr_t)gsrc;
    u32x4 g0;
    g0[0] = 1u;                                        // count=1, user mode
    g0[1] = lds_off;                                   // LDS byte address
    g0[2] = (unsigned)(ga & 0xFFFFFFFFu);              // global_addr[31:0]
    g0[3] = (unsigned)((ga >> 32) & 0x01FFFFFFu)       // global_addr[56:32]
          | 0x80000000u;                               // type=2 ("image")
    unsigned long long s0 = ld;                        // tensor_dim0_stride
    unsigned td0 = ld;                                 // tensor_dim0
    unsigned td1 = rows;                               // tensor_dim1
    i32x8 g1;
    g1[0] = (int)(1u << 16);                           // data_size=1 (2 bytes)
    g1[1] = (int)((td0 & 0xFFFFu) << 16);              // tensor_dim0[15:0]
    g1[2] = (int)((td0 >> 16) | ((td1 & 0xFFFFu) << 16));
    g1[3] = (int)((td1 >> 16) | (klen << 16));         // tile_dim0 = klen
    g1[4] = (int)rows;                                 // tile_dim1; tile_dim2=0
    g1[5] = (int)(s0 & 0xFFFFFFFFull);
    g1[6] = (int)((s0 >> 32) & 0xFFFFull);             // stride1 = 0 (2D)
    g1[7] = 0;
    i32x4 z4 = {0, 0, 0, 0};
#if defined(__clang_major__) && (__clang_major__ >= 23)
    i32x8 z8 = {0, 0, 0, 0, 0, 0, 0, 0};
    __builtin_amdgcn_tensor_load_to_lds(g0, g1, z4, z4, z8, 0);
#else
    __builtin_amdgcn_tensor_load_to_lds(g0, g1, z4, z4, 0);
#endif
}
#endif

// Fallback cooperative staging (256 threads): rows x klen f16, packed.
__device__ inline void stage_panel(const _Float16* src, int ld,
                                   _Float16* dst, int rows, int klen, int tid) {
    const int total = rows * klen;               // klen % 8 == 0
    for (int i = tid * 8; i < total; i += 256 * 8) {
        int r = i / klen;
        int c = i - r * klen;
        *(v8h*)(dst + i) = *(const v8h*)(src + (size_t)r * ld + c);
    }
}

// ---------------------------------------------------------------------------
// Weight convert + transpose: W (K x N f32) -> Wt (N x Kpad f16), zero pad.
// ---------------------------------------------------------------------------
__global__ __launch_bounds__(256) void convert_wt_kernel(
    const float* __restrict__ W, _Float16* __restrict__ Wt,
    int K, int N, int Kpad) {
    int tid = blockIdx.x * 256 + threadIdx.x;
    if (tid >= N * Kpad) return;
    int n = tid / Kpad;
    int k = tid - n * Kpad;
    Wt[tid] = (k < K) ? (_Float16)W[(size_t)k * N + n] : (_Float16)0.0f;
}

// dense_x (B x 13 f32) -> A0 (B x 32 f16, zero padded)
__global__ __launch_bounds__(256) void pad_dense_kernel(
    const float* __restrict__ X, _Float16* __restrict__ A0) {
    int tid = blockIdx.x * 256 + threadIdx.x;
    int b = tid >> 5;
    int k = tid & 31;
    if (b >= BATCH) return;
    A0[tid] = (k < 13) ? (_Float16)X[b * 13 + k] : (_Float16)0.0f;
}

// ---------------------------------------------------------------------------
// LDS-blocked GEMM + bias + ReLU: C = relu(A @ W + b), f16 in/out, f32 acc.
// Workgroup tile 128(M) x 128(N), K-panels of 64 staged in LDS (TDM if
// available). 8 waves; each wave computes a 32x64 subtile (2x4 WMMA frags).
// ~64 FLOP per global byte at the block level.
// ---------------------------------------------------------------------------
__global__ __launch_bounds__(256) void gemm_relu_wmma_kernel(
    const _Float16* __restrict__ A,
    const _Float16* __restrict__ Wt,
    const float*    __restrict__ bias,
    _Float16*       __restrict__ C,
    int Kpad, int ldc, int colOff, int nb /* N/128 */) {
    __shared__ _Float16 Ash[128 * 64];   // 16KB
    __shared__ _Float16 Bsh[128 * 64];   // 16KB

    const int tid  = threadIdx.x;
    const int lane = tid & 31;
    const int wave = tid >> 5;
    const int mb   = blockIdx.x / nb;
    const int nbk  = blockIdx.x - mb * nb;
    const int msub = wave >> 1;          // 0..3 -> rows msub*32 .. +31
    const int nsub = wave & 1;           // 0..1 -> cols nsub*64 .. +63

    const _Float16* Ablk = A  + (size_t)(mb  * 128) * Kpad;
    const _Float16* Bblk = Wt + (size_t)(nbk * 128) * Kpad;

    v8f acc[2][4];
#pragma unroll
    for (int i = 0; i < 2; ++i)
#pragma unroll
        for (int t = 0; t < 4; ++t) acc[i][t] = (v8f){};

    for (int kb = 0; kb < Kpad; kb += 64) {
        const int klen = (Kpad - kb < 64) ? (Kpad - kb) : 64;   // 32 or 64
#ifdef HAVE_TDM
        if (tid < 32) {      // wave 0 drives the DMA; EXEC ignored by TDM
            unsigned aoff = (unsigned)(unsigned long long)(uintptr_t)(void*)Ash;
            unsigned boff = (unsigned)(unsigned long long)(uintptr_t)(void*)Bsh;
            tdm_load_2d(Ablk + kb, aoff, (unsigned)klen, 128u, (unsigned)Kpad);
            tdm_load_2d(Bblk + kb, boff, (unsigned)klen, 128u, (unsigned)Kpad);
            __builtin_amdgcn_s_wait_tensorcnt(0);
        }
#else
        stage_panel(Ablk + kb, Kpad, Ash, 128, klen, tid);
        stage_panel(Bblk + kb, Kpad, Bsh, 128, klen, tid);
#endif
        __syncthreads();

        for (int kk = 0; kk < klen; kk += 32) {
            v16h a0 = load_a_frag(Ash + (size_t)(msub * 32)      * klen, klen, kk, lane);
            v16h a1 = load_a_frag(Ash + (size_t)(msub * 32 + 16) * klen, klen, kk, lane);
#pragma unroll
            for (int t = 0; t < 4; ++t) {
                v16h bm = load_b_frag(Bsh, klen, nsub * 64 + t * 16, kk, lane);
                acc[0][t] = __builtin_amdgcn_wmma_f32_16x16x32_f16(
                    false, a0, false, bm, (short)0, acc[0][t], false, false);
                acc[1][t] = __builtin_amdgcn_wmma_f32_16x16x32_f16(
                    false, a1, false, bm, (short)0, acc[1][t], false, false);
            }
        }
        __syncthreads();
    }

    const int n  = lane & 15;
    const int hl = lane >> 4;
#pragma unroll
    for (int mf = 0; mf < 2; ++mf)
#pragma unroll
        for (int t = 0; t < 4; ++t) {
            const int col = nbk * 128 + nsub * 64 + t * 16 + n;
            const float bv = bias[col];
#pragma unroll
            for (int j = 0; j < 8; ++j) {
                const int row = mb * 128 + msub * 32 + mf * 16 + hl * 8 + j;
                float v = acc[mf][t][j] + bv;
                v = fmaxf(v, 0.0f);
                C[(size_t)row * ldc + colOff + col] = (_Float16)v;
            }
        }
}

// ---------------------------------------------------------------------------
// Interaction: per sample b, T = [x_bot(128) ; 26 embedding rows] (27x128,
// padded to 32) staged as f16 in LDS; Z = T @ T^T via WMMA; strict lower
// triangle -> R[b, 128..478]; pad column R[b,479] zeroed.
// One wave per sample, 8 samples / 256-thread block, 64KB LDS.
// ---------------------------------------------------------------------------
__global__ __launch_bounds__(256) void interact_wmma_kernel(
    const int*   __restrict__ indices,   // (26, B)
    const float* __restrict__ emb,       // (26, VOCAB, 128)
    _Float16*    __restrict__ R) {       // (B, 480): cols 0..127 = x_bot (in)
    __shared__ _Float16 Tsh[8][32 * EMB_D];

    const int wave = threadIdx.x >> 5;
    const int lane = threadIdx.x & 31;
    const int b    = blockIdx.x * 8 + wave;
    _Float16* T = Tsh[wave];

    const _Float16* xrow = R + (size_t)b * 480;
    *(v4h*)&T[0 * EMB_D + lane * 4] = *(const v4h*)(xrow + lane * 4);

#pragma unroll 2
    for (int e = 0; e < N_EMB; ++e) {
        const int idx = indices[e * BATCH + b];
        const float* src = emb + ((size_t)e * VOCAB + (size_t)idx) * EMB_D + lane * 4;
        v4f f = *(const v4f*)src;
        v4h h;
#pragma unroll
        for (int i = 0; i < 4; ++i) h[i] = (_Float16)f[i];
        *(v4h*)&T[(e + 1) * EMB_D + lane * 4] = h;
    }

    v4h z = {};
#pragma unroll
    for (int r = 27; r < 32; ++r) *(v4h*)&T[r * EMB_D + lane * 4] = z;

    if (lane == 0) R[(size_t)b * 480 + 479] = (_Float16)0.0f;

    __syncthreads();

    v8f acc[2][2];
#pragma unroll
    for (int i = 0; i < 2; ++i)
#pragma unroll
        for (int j = 0; j < 2; ++j) acc[i][j] = (v8f){};

    for (int kb = 0; kb < EMB_D; kb += 32) {
        v16h a0 = load_a_frag(T,              EMB_D, kb, lane);
        v16h a1 = load_a_frag(T + 16 * EMB_D, EMB_D, kb, lane);
        v16h b0 = load_b_frag(T, EMB_D, 0,  kb, lane);
        v16h b1 = load_b_frag(T, EMB_D, 16, kb, lane);
        acc[0][0] = __builtin_amdgcn_wmma_f32_16x16x32_f16(false, a0, false, b0, (short)0, acc[0][0], false, false);
        acc[0][1] = __builtin_amdgcn_wmma_f32_16x16x32_f16(false, a0, false, b1, (short)0, acc[0][1], false, false);
        acc[1][0] = __builtin_amdgcn_wmma_f32_16x16x32_f16(false, a1, false, b0, (short)0, acc[1][0], false, false);
        acc[1][1] = __builtin_amdgcn_wmma_f32_16x16x32_f16(false, a1, false, b1, (short)0, acc[1][1], false, false);
    }

    _Float16* zout = R + (size_t)b * 480 + 128;
    const int n  = lane & 15;
    const int hl = lane >> 4;
#pragma unroll
    for (int mt = 0; mt < 2; ++mt)
#pragma unroll
        for (int nt = 0; nt < 2; ++nt)
#pragma unroll
            for (int j = 0; j < 8; ++j) {
                const int ig = mt * 16 + hl * 8 + j;
                const int jg = nt * 16 + n;
                if (ig < 27 && jg < ig)
                    zout[(ig * (ig - 1)) / 2 + jg] = (_Float16)acc[mt][nt][j];
            }
}

// ---------------------------------------------------------------------------
// Final layer: out[b] = clip(sigmoid(T3[b,:].w + bias), 0, 1). One wave/row.
// ---------------------------------------------------------------------------
__global__ __launch_bounds__(256) void gemv_sigmoid_kernel(
    const _Float16* __restrict__ T3,
    const _Float16* __restrict__ w,
    const float*    __restrict__ bias,
    float*          __restrict__ out) {
    const int wave = (blockIdx.x * 256 + threadIdx.x) >> 5;
    const int lane = threadIdx.x & 31;
    if (wave >= BATCH) return;

    v16h x  = *(const v16h*)(T3 + (size_t)wave * 512 + lane * 16);
    v16h wv = *(const v16h*)(w + lane * 16);
    float s = 0.0f;
#pragma unroll
    for (int i = 0; i < 16; ++i) s += (float)x[i] * (float)wv[i];
#pragma unroll
    for (int off = 16; off > 0; off >>= 1) s += __shfl_down(s, off, 32);
    if (lane == 0) {
        float v = s + bias[0];
        float p = 1.0f / (1.0f + __expf(-v));
        out[wave] = fminf(fmaxf(p, 0.0f), 1.0f);
    }
}

// ---------------------------------------------------------------------------
// Host launcher
// ---------------------------------------------------------------------------
extern "C" void kernel_launch(void* const* d_in, const int* in_sizes, int n_in,
                              void* d_out, int out_size, void* d_ws, size_t ws_size,
                              hipStream_t stream) {
    (void)in_sizes; (void)n_in; (void)out_size; (void)ws_size;

    const float* dense_x = (const float*)d_in[0];
    const int*   indices = (const int*)  d_in[1];
    const float* emb     = (const float*)d_in[2];
    const float* botW[3] = { (const float*)d_in[3], (const float*)d_in[5], (const float*)d_in[7] };
    const float* botB[3] = { (const float*)d_in[4], (const float*)d_in[6], (const float*)d_in[8] };
    const float* topW[4] = { (const float*)d_in[9], (const float*)d_in[11], (const float*)d_in[13], (const float*)d_in[15] };
    const float* topB[4] = { (const float*)d_in[10], (const float*)d_in[12], (const float*)d_in[14], (const float*)d_in[16] };
    float* out = (float*)d_out;

    char* base = (char*)d_ws;
    size_t off = 0;
    auto carve = [&](size_t bytes) -> char* {
        char* p = base + off;
        off = (off + bytes + 255) & ~(size_t)255;
        return p;
    };
    _Float16* wt_b0 = (_Float16*)carve((size_t)512  * 32   * 2);
    _Float16* wt_b1 = (_Float16*)carve((size_t)256  * 512  * 2);
    _Float16* wt_b2 = (_Float16*)carve((size_t)128  * 256  * 2);
    _Float16* wt_t0 = (_Float16*)carve((size_t)1024 * 480  * 2);
    _Float16* wt_t1 = (_Float16*)carve((size_t)1024 * 1024 * 2);
    _Float16* wt_t2 = (_Float16*)carve((size_t)512  * 1024 * 2);
    _Float16* wt_t3 = (_Float16*)carve((size_t)512         * 2);
    _Float16* A0    = (_Float16*)carve((size_t)BATCH * 32   * 2);
    _Float16* H1    = (_Float16*)carve((size_t)BATCH * 512  * 2);
    _Float16* H2    = (_Float16*)carve((size_t)BATCH * 256  * 2);
    _Float16* R     = (_Float16*)carve((size_t)BATCH * 480  * 2);
    _Float16* T1    = (_Float16*)carve((size_t)BATCH * 1024 * 2);
    _Float16* T2    = (_Float16*)carve((size_t)BATCH * 1024 * 2);
    _Float16* T3    = (_Float16*)carve((size_t)BATCH * 512  * 2);

    auto cvt = [&](const float* W, _Float16* Wt, int K, int N, int Kpad) {
        int total = N * Kpad;
        convert_wt_kernel<<<(total + 255) / 256, 256, 0, stream>>>(W, Wt, K, N, Kpad);
    };
    cvt(botW[0], wt_b0,   13,  512,   32);
    cvt(botW[1], wt_b1,  512,  256,  512);
    cvt(botW[2], wt_b2,  256,  128,  256);
    cvt(topW[0], wt_t0,  479, 1024,  480);
    cvt(topW[1], wt_t1, 1024, 1024, 1024);
    cvt(topW[2], wt_t2, 1024,  512, 1024);
    cvt(topW[3], wt_t3,  512,    1,  512);

    pad_dense_kernel<<<(BATCH * 32) / 256, 256, 0, stream>>>(dense_x, A0);

    auto gemm = [&](const _Float16* A, const _Float16* Wt, const float* b,
                    _Float16* C, int Kpad, int N, int ldc, int colOff) {
        int nb = N / 128;
        gemm_relu_wmma_kernel<<<(BATCH / 128) * nb, 256, 0, stream>>>(
            A, Wt, b, C, Kpad, ldc, colOff, nb);
    };

    // bottom MLP: 13->512->256->128 (last layer lands in R cols 0..127)
    gemm(A0, wt_b0, botB[0], H1,  32, 512, 512, 0);
    gemm(H1, wt_b1, botB[1], H2, 512, 256, 256, 0);
    gemm(H2, wt_b2, botB[2], R,  256, 128, 480, 0);

    // pairwise interactions -> R cols 128..478 (+ zero pad col 479)
    interact_wmma_kernel<<<BATCH / 8, 256, 0, stream>>>(indices, emb, R);

    // top MLP: 479(->480) -> 1024 -> 1024 -> 512
    gemm(R,  wt_t0, topB[0], T1,  480, 1024, 1024, 0);
    gemm(T1, wt_t1, topB[1], T2, 1024, 1024, 1024, 0);
    gemm(T2, wt_t2, topB[2], T3, 1024,  512,  512, 0);

    // 512 -> 1 with sigmoid + clip
    gemv_sigmoid_kernel<<<(BATCH * 32) / 256, 256, 0, stream>>>(T3, wt_t3, topB[3], out);
}
